// ScaledDotProductAttention_7026566496615
// MI455X (gfx1250) — matrix operand order repaired and checked by reading, not compiled, not run on hardware
//
#include <hip/hip_runtime.h>

typedef __attribute__((ext_vector_type(16))) __bf16 v16bf;
typedef __attribute__((ext_vector_type(8)))  __bf16 v8bf;
typedef __attribute__((ext_vector_type(8)))  float  v8f;

#define B_   16
#define S_   2048
#define D_   128
#define SCALE 0.08838834764831845f   // 1/sqrt(128)

union AFrag { v16bf v; v8bf h[2]; };

// 8x f32 -> 8x bf16 (lowers to v_cvt_pk_bf16_f32 x4)
__device__ inline v8bf cvt8(float4 a, float4 b) {
  v8f f;
  f[0] = a.x; f[1] = a.y; f[2] = a.z; f[3] = a.w;
  f[4] = b.x; f[5] = b.y; f[6] = b.z; f[7] = b.w;
  return __builtin_convertvector(f, v8bf);
}

// scaled variant (used once for Q; folds 1/sqrt(D) into the A operand)
__device__ inline v8bf cvt8s(float4 a, float4 b, float s) {
  v8f f;
  f[0] = a.x * s; f[1] = a.y * s; f[2] = a.z * s; f[3] = a.w * s;
  f[4] = b.x * s; f[5] = b.y * s; f[6] = b.z * s; f[7] = b.w * s;
  return __builtin_convertvector(f, v8bf);
}

__device__ inline unsigned pk2(float a, float b) {
  union { __bf16 h[2]; unsigned u; } t;
  t.h[0] = (__bf16)a; t.h[1] = (__bf16)b;
  return t.u;
}

// ---------------------------------------------------------------------------
// Pre-pass 1: K (f32, row-major) -> bf16, same layout. 8 elems/thread.
// ---------------------------------------------------------------------------
__global__ __launch_bounds__(256)
void convert_bf16_kernel(const float* __restrict__ src, unsigned* __restrict__ dst) {
  const size_t i = ((size_t)blockIdx.x * 256 + threadIdx.x) * 8;
  const float4* s4 = (const float4*)(src + i);
  float4 a = s4[0], b = s4[1];
  uint4 o;
  o.x = pk2(a.x, a.y); o.y = pk2(a.z, a.w);
  o.z = pk2(b.x, b.y); o.w = pk2(b.z, b.w);
  *(uint4*)(dst + i / 2) = o;
}

// ---------------------------------------------------------------------------
// Pre-pass 2: V[b][k][n] (f32) -> Vt[b][n][k] (bf16). 32x32 LDS tile.
// ---------------------------------------------------------------------------
__global__ __launch_bounds__(256)
void transpose_v_bf16_kernel(const float* __restrict__ V, unsigned* __restrict__ Vt) {
  __shared__ float t[32][33];
  const int k0 = blockIdx.x * 32;
  const int n0 = blockIdx.y * 32;
  const int b  = blockIdx.z;
  const int tx = threadIdx.x & 31;
  const int ty = threadIdx.x >> 5;            // 0..7
  const float* src = V + ((size_t)b * S_ + k0) * D_ + n0;
  #pragma unroll
  for (int j = 0; j < 4; ++j)
    t[ty + 8 * j][tx] = src[(size_t)(ty + 8 * j) * D_ + tx];
  __syncthreads();
  const int n  = threadIdx.x >> 4;            // 0..15
  const int kp = threadIdx.x & 15;            // dword (k-pair) 0..15
  unsigned* dst = Vt + ((((size_t)b * D_ + n0) * S_) + k0) / 2;
  #pragma unroll
  for (int j = 0; j < 2; ++j) {
    const int nn = n + 16 * j;
    dst[(size_t)nn * (S_ / 2) + kp] = pk2(t[2 * kp][nn], t[2 * kp + 1][nn]);
  }
}

// ---------------------------------------------------------------------------
// Fast fused attention: Q f32, K bf16 (row-major), Vt bf16 (transposed)
// ---------------------------------------------------------------------------
__global__ __launch_bounds__(256)
void attn_fast_kernel(const float* __restrict__ Q,
                      const unsigned short* __restrict__ Kbf,
                      const unsigned short* __restrict__ Vt,
                      float* __restrict__ out,
                      float* __restrict__ attn) {
  __shared__ float    sS[16 * S_];       // 128 KB f32 scores -> exp
  __shared__ unsigned sE[16 * S_ / 2];   //  64 KB packed bf16 exp
  __shared__ float    red[256];
  __shared__ float    rowmax[16];
  __shared__ float    rowinv[16];

  const int qt   = blockIdx.x;
  const int b    = blockIdx.y;
  const int tid  = threadIdx.x;
  const int wave = tid >> 5;
  const int lane = tid & 31;
  const int mrow = lane & 15;
  const int hi   = (lane >> 4) & 1;
  const int hi8  = hi ? 8 : 0;
  const int hi16 = hi ? 16 : 0;

  const size_t bS = (size_t)b * S_;
  const int q0 = qt * 16;

  // Q A-fragments (built once; f32*scale -> bf16)
  AFrag qa[4];
  {
    const float* qrow = Q + (bS + q0 + mrow) * D_;
    #pragma unroll
    for (int c = 0; c < 4; ++c) {
      const int base = c * 32 + hi16;
      qa[c].h[0] = cvt8s(*(const float4*)(qrow + base),
                         *(const float4*)(qrow + base + 4), SCALE);
      qa[c].h[1] = cvt8s(*(const float4*)(qrow + base + 8),
                         *(const float4*)(qrow + base + 12), SCALE);
    }
  }

  // S = (Q*scale) K^T; constant trip count -> no exec-mask loop, full unroll ok
  #pragma unroll 2
  for (int t = 0; t < S_ / 16 / 8; ++t) {
    const int kt = wave + t * 8;
    const __bf16* krow = (const __bf16*)Kbf + (bS + kt * 16 + mrow) * D_ + hi16;
    v8f acc = {};
    #pragma unroll
    for (int c = 0; c < 4; ++c) {
      AFrag kb;
      kb.h[0] = *(const v8bf*)(krow + c * 32);
      kb.h[1] = *(const v8bf*)(krow + c * 32 + 8);
      acc = __builtin_amdgcn_wmma_f32_16x16x32_bf16(
          false, qa[c].v, false, kb.v, (short)0, acc, false, false);
    }
    #pragma unroll
    for (int i = 0; i < 8; ++i)
      sS[(i + 8 * hi) * S_ + kt * 16 + mrow] = acc[i];
  }
  __syncthreads();

  // Softmax: 16 threads per row
  {
    const int r  = tid >> 4;
    const int c0 = (tid & 15) * 128;
    float4* seg4 = (float4*)(sS + r * S_ + c0);

    float m = -3.402823466e38f;
    #pragma unroll 4
    for (int j = 0; j < 32; ++j) {
      float4 x = seg4[j];
      m = fmaxf(m, fmaxf(fmaxf(x.x, x.y), fmaxf(x.z, x.w)));
    }
    red[tid] = m;
    __syncthreads();
    if ((tid & 15) == 0) {
      float mm = red[tid];
      #pragma unroll
      for (int j = 1; j < 16; ++j) mm = fmaxf(mm, red[tid + j]);
      rowmax[r] = mm;
    }
    __syncthreads();

    const float mm = rowmax[r];
    unsigned* e32 = sE + (r * S_ + c0) / 2;
    float ssum = 0.f;
    #pragma unroll 4
    for (int j = 0; j < 32; ++j) {
      float4 x = seg4[j];
      float4 e;
      e.x = __expf(x.x - mm); e.y = __expf(x.y - mm);
      e.z = __expf(x.z - mm); e.w = __expf(x.w - mm);
      seg4[j] = e;
      e32[2 * j]     = pk2(e.x, e.y);
      e32[2 * j + 1] = pk2(e.z, e.w);
      ssum += (e.x + e.y) + (e.z + e.w);
    }
    red[tid] = ssum;
    __syncthreads();
    if ((tid & 15) == 0) {
      float s = 0.f;
      #pragma unroll
      for (int j = 0; j < 16; ++j) s += red[tid + j];
      rowinv[r] = 1.0f / s;
    }
    __syncthreads();
  }

  // attn = E * rowinv (coalesced float4)
  {
    float4* attn4 = (float4*)(attn + (bS + q0) * (size_t)S_);
    const float4* s4 = (const float4*)sS;
    for (int idx4 = tid; idx4 < 16 * S_ / 4; idx4 += 256) {
      const float inv = rowinv[idx4 >> 9];
      float4 v = s4[idx4];
      v.x *= inv; v.y *= inv; v.z *= inv; v.w *= inv;
      attn4[idx4] = v;
    }
  }

  // O = diag(rowinv) * E * V; Vt transposed bf16: pure b128 loads
  {
    const int ncol = wave * 16 + mrow;
    const __bf16* prow0 = (const __bf16*)sE + mrow * S_ + hi16;
    const __bf16* vrow  = (const __bf16*)Vt + ((size_t)b * D_ + ncol) * S_ + hi16;
    v8f acc = {};
    #pragma unroll 2
    for (int k0 = 0; k0 < S_; k0 += 32) {
      AFrag pa;
      pa.h[0] = *(const v8bf*)(prow0 + k0);
      pa.h[1] = *(const v8bf*)(prow0 + k0 + 8);
      AFrag vb;
      vb.h[0] = *(const v8bf*)(vrow + k0);
      vb.h[1] = *(const v8bf*)(vrow + k0 + 8);
      acc = __builtin_amdgcn_wmma_f32_16x16x32_bf16(
          false, pa.v, false, vb.v, (short)0, acc, false, false);
    }
    float* obase = out + (bS + q0) * D_ + ncol;
    #pragma unroll
    for (int i = 0; i < 8; ++i) {
      const int m = i + 8 * hi;
      obase[(size_t)m * D_] = acc[i] * rowinv[m];
    }
  }
}

// ---------------------------------------------------------------------------
// Fallback fused kernel (no workspace): converts K/V inline.
// ---------------------------------------------------------------------------
__global__ __launch_bounds__(256)
void attn_fused_kernel(const float* __restrict__ Q,
                       const float* __restrict__ K,
                       const float* __restrict__ V,
                       float* __restrict__ out,
                       float* __restrict__ attn) {
  __shared__ float    sS[16 * S_];
  __shared__ unsigned sE[16 * S_ / 2];
  __shared__ float    red[256];
  __shared__ float    rowmax[16];
  __shared__ float    rowinv[16];

  const int qt   = blockIdx.x;
  const int b    = blockIdx.y;
  const int tid  = threadIdx.x;
  const int wave = tid >> 5;
  const int lane = tid & 31;
  const int mrow = lane & 15;
  const int hi   = (lane >> 4) & 1;
  const int hi8  = hi ? 8 : 0;

  const size_t bS = (size_t)b * S_;
  const int q0 = qt * 16;

  AFrag qa[4];
  {
    const float* qrow = Q + (bS + q0 + mrow) * D_;
    #pragma unroll
    for (int c = 0; c < 4; ++c) {
      const int base = c * 32 + (hi ? 16 : 0);
      qa[c].h[0] = cvt8s(*(const float4*)(qrow + base),
                         *(const float4*)(qrow + base + 4), SCALE);
      qa[c].h[1] = cvt8s(*(const float4*)(qrow + base + 8),
                         *(const float4*)(qrow + base + 12), SCALE);
    }
  }

  #pragma unroll 2
  for (int t = 0; t < S_ / 16 / 8; ++t) {
    const int kt = wave + t * 8;
    const float* krow = K + (bS + kt * 16 + mrow) * D_ + (hi ? 16 : 0);
    v8f acc = {};
    #pragma unroll
    for (int c = 0; c < 4; ++c) {
      const float4* kp = (const float4*)(krow + c * 32);
      AFrag kb;
      kb.h[0] = cvt8(kp[0], kp[1]);
      kb.h[1] = cvt8(kp[2], kp[3]);
      acc = __builtin_amdgcn_wmma_f32_16x16x32_bf16(
          false, qa[c].v, false, kb.v, (short)0, acc, false, false);
    }
    #pragma unroll
    for (int i = 0; i < 8; ++i)
      sS[(i + 8 * hi) * S_ + kt * 16 + mrow] = acc[i];
  }
  __syncthreads();

  {
    const int r  = tid >> 4;
    const int c0 = (tid & 15) * 128;
    float4* seg4 = (float4*)(sS + r * S_ + c0);
    float m = -3.402823466e38f;
    #pragma unroll 4
    for (int j = 0; j < 32; ++j) {
      float4 x = seg4[j];
      m = fmaxf(m, fmaxf(fmaxf(x.x, x.y), fmaxf(x.z, x.w)));
    }
    red[tid] = m;
    __syncthreads();
    if ((tid & 15) == 0) {
      float mm = red[tid];
      #pragma unroll
      for (int j = 1; j < 16; ++j) mm = fmaxf(mm, red[tid + j]);
      rowmax[r] = mm;
    }
    __syncthreads();
    const float mm = rowmax[r];
    unsigned* e32 = sE + (r * S_ + c0) / 2;
    float ssum = 0.f;
    #pragma unroll 4
    for (int j = 0; j < 32; ++j) {
      float4 x = seg4[j];
      float4 e;
      e.x = __expf(x.x - mm); e.y = __expf(x.y - mm);
      e.z = __expf(x.z - mm); e.w = __expf(x.w - mm);
      seg4[j] = e;
      e32[2 * j]     = pk2(e.x, e.y);
      e32[2 * j + 1] = pk2(e.z, e.w);
      ssum += (e.x + e.y) + (e.z + e.w);
    }
    red[tid] = ssum;
    __syncthreads();
    if ((tid & 15) == 0) {
      float s = 0.f;
      #pragma unroll
      for (int j = 0; j < 16; ++j) s += red[tid + j];
      rowinv[r] = 1.0f / s;
    }
    __syncthreads();
  }

  {
    float4* attn4 = (float4*)(attn + (bS + q0) * (size_t)S_);
    const float4* s4 = (const float4*)sS;
    for (int idx4 = tid; idx4 < 16 * S_ / 4; idx4 += 256) {
      const float inv = rowinv[idx4 >> 9];
      float4 v = s4[idx4];
      v.x *= inv; v.y *= inv; v.z *= inv; v.w *= inv;
      attn4[idx4] = v;
    }
  }

  {
    const int ncol = wave * 16 + mrow;
    const float* vcol = V + bS * D_ + ncol;
    const __bf16* prow0 = (const __bf16*)sE + mrow * S_ + (hi ? 16 : 0);
    v8f acc = {};
    #pragma unroll 2
    for (int k0 = 0; k0 < S_; k0 += 32) {
      AFrag pa;
      pa.h[0] = *(const v8bf*)(prow0 + k0);
      pa.h[1] = *(const v8bf*)(prow0 + k0 + 8);
      const float* vk = vcol + (size_t)(k0 + (hi ? 16 : 0)) * D_;
      float4 f0, f1, f2, f3;
      f0.x = vk[0 * D_];  f0.y = vk[1 * D_];  f0.z = vk[2 * D_];  f0.w = vk[3 * D_];
      f1.x = vk[4 * D_];  f1.y = vk[5 * D_];  f1.z = vk[6 * D_];  f1.w = vk[7 * D_];
      f2.x = vk[8 * D_];  f2.y = vk[9 * D_];  f2.z = vk[10 * D_]; f2.w = vk[11 * D_];
      f3.x = vk[12 * D_]; f3.y = vk[13 * D_]; f3.z = vk[14 * D_]; f3.w = vk[15 * D_];
      AFrag vb;
      vb.h[0] = cvt8(f0, f1);
      vb.h[1] = cvt8(f2, f3);
      acc = __builtin_amdgcn_wmma_f32_16x16x32_bf16(
          false, pa.v, false, vb.v, (short)0, acc, false, false);
    }
    float* obase = out + (bS + q0) * D_ + ncol;
    #pragma unroll
    for (int i = 0; i < 8; ++i) {
      const int m = i + 8 * hi;
      obase[(size_t)m * D_] = acc[i] * rowinv[m];
    }
  }
}

extern "C" void kernel_launch(void* const* d_in, const int* in_sizes, int n_in,
                              void* d_out, int out_size, void* d_ws, size_t ws_size,
                              hipStream_t stream) {
  (void)in_sizes; (void)n_in; (void)out_size;
  const float* Q = (const float*)d_in[0];
  const float* K = (const float*)d_in[1];
  const float* V = (const float*)d_in[2];
  float* out  = (float*)d_out;                          // [B,S,D]
  float* attn = (float*)d_out + (size_t)B_ * S_ * D_;   // [B,S,S]
  dim3 grid(S_ / 16, B_);

  const size_t NE = (size_t)B_ * S_ * D_;               // elements per tensor
  if (ws_size >= NE * 4) {                              // Kbf16 + Vt bf16 = 16 MB
    unsigned short* Kbf = (unsigned short*)d_ws;
    unsigned short* Vt  = (unsigned short*)d_ws + NE;
    convert_bf16_kernel<<<(unsigned)(NE / (256 * 8)), 256, 0, stream>>>(
        K, (unsigned*)Kbf);
    transpose_v_bf16_kernel<<<dim3(S_ / 32, D_ / 32, B_), 256, 0, stream>>>(
        V, (unsigned*)Vt);
    attn_fast_kernel<<<grid, 256, 0, stream>>>(Q, Kbf, Vt, out, attn);
  } else {
    attn_fused_kernel<<<grid, 256, 0, stream>>>(Q, K, V, out, attn);
  }
}
